// TransformerEncoder_13005160972421
// MI455X (gfx1250) — matrix operand order repaired and checked by reading, not compiled, not run on hardware
//
#include <hip/hip_runtime.h>
#include <stdint.h>
#include <stddef.h>

typedef unsigned short u16;
typedef __attribute__((ext_vector_type(16))) __bf16 bf16x16;
typedef __attribute__((ext_vector_type(8)))  float  f32x8;

#define DEV static __device__ __forceinline__

namespace {
constexpr int B_ = 8, T_ = 1024, D_ = 1024, H_ = 16, L_ = 6;
constexpr int HS_ = 64, FF_ = 4096, NH_ = 4096, NO_ = 1000;

DEV u16 f2bf(float f) {
  union { float f; uint32_t u; } v; v.f = f;
  uint32_t u = v.u;
  return (u16)((u + 0x7FFFu + ((u >> 16) & 1u)) >> 16);  // RNE
}
DEV float bf2f(u16 h) {
  union { uint32_t u; float f; } v; v.u = ((uint32_t)h) << 16;
  return v.f;
}
DEV f32x8 zero8() {
  f32x8 v = {0.f, 0.f, 0.f, 0.f, 0.f, 0.f, 0.f, 0.f};
  return v;
}
// elems 0..7 from p0, elems 8..15 from p1 (each 16B, 16B-aligned)
DEV bf16x16 ldfrag(const u16* p0, const u16* p1) {
  bf16x16 f;
  ((uint4*)&f)[0] = *(const uint4*)p0;
  ((uint4*)&f)[1] = *(const uint4*)p1;
  return f;
}
DEV f32x8 wmma_bf(bf16x16 a, bf16x16 b, f32x8 c) {
  return __builtin_amdgcn_wmma_f32_16x16x32_bf16(false, a, false, b, (short)0, c,
                                                 false, false);
}

// ---- DPP16 cross-lane reductions over 16 consecutive lanes (no LDS traffic)
template <int CTRL>
DEV float dpp_movf(float x) {
  return __int_as_float(
      __builtin_amdgcn_update_dpp(0, __float_as_int(x), CTRL, 0xf, 0xf, true));
}
DEV float red16_add(float x) {
  x += dpp_movf<0xB1>(x);    // quad_perm [1,0,3,2]  : xor 1
  x += dpp_movf<0x4E>(x);    // quad_perm [2,3,0,1]  : xor 2
  x += dpp_movf<0x141>(x);   // row_half_mirror      : combine quads
  x += dpp_movf<0x140>(x);   // row_mirror           : combine octets
  return x;
}
DEV float red16_max(float x) {
  x = fmaxf(x, dpp_movf<0xB1>(x));
  x = fmaxf(x, dpp_movf<0x4E>(x));
  x = fmaxf(x, dpp_movf<0x141>(x));
  x = fmaxf(x, dpp_movf<0x140>(x));
  return x;
}
}  // namespace

// ---------------------------------------------------------------- utilities
__global__ __launch_bounds__(256) void k_f32_to_bf16(const float* __restrict__ s,
                                                     u16* __restrict__ d, size_t n) {
  size_t i = (size_t)blockIdx.x * 256 + threadIdx.x;
  if (i < n) d[i] = f2bf(s[i]);
}

__global__ __launch_bounds__(256) void k_embed(const int* __restrict__ x,
                                               const float* __restrict__ emb,
                                               const float* __restrict__ pos,
                                               float* __restrict__ h) {
  int row = blockIdx.x;  // b*T + t
  int t = row & (T_ - 1);
  int tok = x[row];
  const float* e = emb + (size_t)tok * D_;
  const float* p = pos + (size_t)t * D_;
  float* hr = h + (size_t)row * D_;
#pragma unroll
  for (int i = 0; i < 4; i++) {
    int c = threadIdx.x + i * 256;
    hr[c] = e[c] + p[c];
  }
}

__global__ __launch_bounds__(256) void k_layernorm(const float* __restrict__ x,
                                                   const float* __restrict__ g,
                                                   const float* __restrict__ bta,
                                                   u16* __restrict__ z) {
  __shared__ float red[8];
  const int row = blockIdx.x, tid = threadIdx.x, lane = tid & 31, w = tid >> 5;
  const float* xr = x + (size_t)row * D_;
  float v[4];
  float s = 0.f;
#pragma unroll
  for (int i = 0; i < 4; i++) { v[i] = xr[tid + i * 256]; s += v[i]; }
  s = red16_add(s);
  s += __shfl_xor(s, 16, 32);
  if (lane == 0) red[w] = s;
  __syncthreads();
  float tot = 0.f;
#pragma unroll
  for (int i = 0; i < 8; i++) tot += red[i];
  float mu = tot * (1.f / D_);
  __syncthreads();
  float q = 0.f;
#pragma unroll
  for (int i = 0; i < 4; i++) { float d = v[i] - mu; q += d * d; }
  q = red16_add(q);
  q += __shfl_xor(q, 16, 32);
  if (lane == 0) red[w] = q;
  __syncthreads();
  float tq = 0.f;
#pragma unroll
  for (int i = 0; i < 8; i++) tq += red[i];
  float r = rsqrtf(tq * (1.f / D_) + 1e-5f);
#pragma unroll
  for (int i = 0; i < 4; i++) {
    int c = tid + i * 256;
    z[(size_t)row * D_ + c] = f2bf((v[i] - mu) * r * g[c] + bta[c]);
  }
}

// ---------------------------------------------------------------- WMMA GEMM
// C[M,N] = epilogue(A[M,K](bf16) x B[K,N](bf16)); 128x128 tile, K-stage 64,
// register-staged software pipeline (prefetch next tile during compute).
template <bool BIAS, bool RELU, bool RESID, bool BF16OUT, bool SCALE>
__global__ __launch_bounds__(256) void k_gemm(const u16* __restrict__ A,
                                              const u16* __restrict__ Bw,
                                              const float* __restrict__ bias,
                                              const float* __restrict__ resid,
                                              void* __restrict__ Cout,
                                              int M, int N, int K, float oscale) {
  __shared__ __align__(16) u16 As[128][72];   // [m][k], row 144B
  __shared__ __align__(16) u16 Bs[128][72];   // transposed: [n][k]
  const int tid = threadIdx.x, lane = tid & 31, w = tid >> 5;
  const int wr = w >> 1, wc = w & 1;
  const int m0 = blockIdx.y * 128, n0 = blockIdx.x * 128;
  const int l15 = lane & 15, kh = lane >> 4;

  f32x8 acc[2][4];
#pragma unroll
  for (int i = 0; i < 2; i++)
#pragma unroll
    for (int j = 0; j < 4; j++) acc[i][j] = zero8();

  const int arow = tid >> 1, acol = (tid & 1) * 32;  // 128x64 A tile
  // B tile 64x128: each thread owns k-rows {bkp, bkp+1} x 16 n-cols
  const int bkp = (tid & 31) * 2, bn = (tid >> 5) * 16;

  uint4 ra[4], rb[4];
  {  // prologue: prefetch tile 0 into registers
    const u16* asrc = A + (size_t)(m0 + arow) * K + acol;
    const u16* b0 = Bw + (size_t)bkp * N + n0 + bn;
    const u16* b1 = b0 + N;
#pragma unroll
    for (int i = 0; i < 4; i++) ra[i] = ((const uint4*)asrc)[i];
    rb[0] = ((const uint4*)b0)[0]; rb[1] = ((const uint4*)b0)[1];
    rb[2] = ((const uint4*)b1)[0]; rb[3] = ((const uint4*)b1)[1];
  }

  for (int k0 = 0; k0 < K; k0 += 64) {
    // stage registers -> LDS
#pragma unroll
    for (int i = 0; i < 4; i++) *(uint4*)&As[arow][acol + 8 * i] = ra[i];
    {
      u16 t0[16], t1[16];
      ((uint4*)t0)[0] = rb[0]; ((uint4*)t0)[1] = rb[1];
      ((uint4*)t1)[0] = rb[2]; ((uint4*)t1)[1] = rb[3];
#pragma unroll
      for (int e = 0; e < 16; e++)   // packed {k,k+1} dwords, conflict-free
        *(uint32_t*)&Bs[bn + e][bkp] =
            (uint32_t)t0[e] | ((uint32_t)t1[e] << 16);
    }
    __syncthreads();

    // prefetch next tile; latency hidden behind the 16 WMMAs below
    if (k0 + 64 < K) {
      const u16* asrc = A + (size_t)(m0 + arow) * K + (k0 + 64) + acol;
      const u16* b0 = Bw + (size_t)(k0 + 64 + bkp) * N + n0 + bn;
      const u16* b1 = b0 + N;
#pragma unroll
      for (int i = 0; i < 4; i++) ra[i] = ((const uint4*)asrc)[i];
      rb[0] = ((const uint4*)b0)[0]; rb[1] = ((const uint4*)b0)[1];
      rb[2] = ((const uint4*)b1)[0]; rb[3] = ((const uint4*)b1)[1];
    }

#pragma unroll
    for (int kc = 0; kc < 2; kc++) {
      bf16x16 af[2], bfr[4];
#pragma unroll
      for (int mi = 0; mi < 2; mi++) {
        int r = wr * 32 + mi * 16 + l15;
        af[mi] = ldfrag(&As[r][kc * 32 + kh * 8], &As[r][kc * 32 + 16 + kh * 8]);
      }
#pragma unroll
      for (int ni = 0; ni < 4; ni++) {
        int c = wc * 64 + ni * 16 + l15;
        bfr[ni] = ldfrag(&Bs[c][kc * 32 + kh * 8], &Bs[c][kc * 32 + 16 + kh * 8]);
      }
#pragma unroll
      for (int mi = 0; mi < 2; mi++)
#pragma unroll
        for (int ni = 0; ni < 4; ni++)
          acc[mi][ni] = wmma_bf(af[mi], bfr[ni], acc[mi][ni]);
    }
    __syncthreads();
  }

  const int rb8 = 8 * kh;
#pragma unroll
  for (int mi = 0; mi < 2; mi++)
#pragma unroll
    for (int ni = 0; ni < 4; ni++)
#pragma unroll
      for (int r = 0; r < 8; r++) {
        int row = m0 + wr * 32 + mi * 16 + rb8 + r;
        int col = n0 + wc * 64 + ni * 16 + l15;
        float vv = acc[mi][ni][r];
        if (SCALE) vv *= oscale;
        if (BIAS) vv += bias[col];
        if (RESID) vv += resid[(size_t)row * N + col];
        if (RELU) vv = fmaxf(vv, 0.f);
        if (BF16OUT)
          ((u16*)Cout)[(size_t)row * N + col] = f2bf(vv);
        else
          ((float*)Cout)[(size_t)row * N + col] = vv;
      }
}

// ---------------------------------------------------------- flash attention
// grid (T/128, H, B), 256 threads (8 waves); wave owns 16 query rows.
// Q is pre-scaled by D^-0.5 in its producing GEMM. Online softmax in the
// WMMA C layout with DPP16 row reductions; P round-trips LDS (wave-private;
// same-wave DS ordering, no block barrier). Next key tile is warmed via
// global_prefetch (zero VGPR cost) instead of register prefetch, so the
// staging loads complete at cache latency without spilling.
__global__ __launch_bounds__(256, 1) void k_flash_attn(const u16* __restrict__ q,
                                                       const u16* __restrict__ k,
                                                       const u16* __restrict__ v,
                                                       u16* __restrict__ o) {
  __shared__ __align__(16) u16 Ks[64][72];      // [tk][hs]
  __shared__ __align__(16) u16 Vt[64][72];      // [hs][tk]
  __shared__ __align__(16) u16 Ps[8][16][72];   // per-wave P tile [m][tk]
  const int tid = threadIdx.x, lane = tid & 31, w = tid >> 5;
  const int l15 = lane & 15, kh = lane >> 4;
  const int bb = blockIdx.z, hh = blockIdx.y;
  const int qt0 = blockIdx.x * 128 + w * 16;

  bf16x16 qf[2];
  {
    const u16* qp = q + ((size_t)(bb * T_ + qt0 + l15) * H_ + hh) * HS_;
#pragma unroll
    for (int c = 0; c < 2; c++)
      qf[c] = ldfrag(qp + c * 32 + kh * 8, qp + c * 32 + 16 + kh * 8);
  }

  float mi[8], li[8];
  f32x8 oacc[4];
#pragma unroll
  for (int r = 0; r < 8; r++) { mi[r] = -3.0e38f; li[r] = 0.f; }
#pragma unroll
  for (int ni = 0; ni < 4; ni++) oacc[ni] = zero8();

  const int krK = tid >> 2, hbK = (tid & 3) * 16;   // K: 16 elems/thread
  const int krV = (tid & 31) * 2, hbV = (tid >> 5) * 8;  // V: 2 rows x 8 cols

  const size_t tstride = (size_t)H_ * HS_;  // elements per token row
  const u16* kbase = k + ((size_t)bb * T_ * H_ + hh) * HS_;
  const u16* vbase = v + ((size_t)bb * T_ * H_ + hh) * HS_;

  for (int kt0 = 0; kt0 < T_; kt0 += 64) {
    {  // stage K (direct) and V (transposed, packed {tk,tk+1} dwords)
      const u16* kp = kbase + (size_t)(kt0 + krK) * tstride + hbK;
      const u16* vp = vbase + (size_t)(kt0 + krV) * tstride + hbV;
      uint4 rk0 = ((const uint4*)kp)[0];
      uint4 rk1 = ((const uint4*)kp)[1];
      uint4 rv0 = *(const uint4*)vp;
      uint4 rv1 = *(const uint4*)(vp + tstride);
      *(uint4*)&Ks[krK][hbK + 0] = rk0;
      *(uint4*)&Ks[krK][hbK + 8] = rk1;
      u16 t0[8], t1[8];
      *(uint4*)t0 = rv0;
      *(uint4*)t1 = rv1;
#pragma unroll
      for (int e = 0; e < 8; e++)
        *(uint32_t*)&Vt[hbV + e][krV] =
            (uint32_t)t0[e] | ((uint32_t)t1[e] << 16);
    }
    __syncthreads();

    if (kt0 + 64 < T_) {  // warm L2/WGP$ for the next key tile (no VGPR cost)
      __builtin_prefetch(kbase + (size_t)(kt0 + 64 + krK) * tstride + hbK, 0, 3);
      __builtin_prefetch(vbase + (size_t)(kt0 + 64 + krV) * tstride + hbV, 0, 3);
      __builtin_prefetch(vbase + (size_t)(kt0 + 65 + krV) * tstride + hbV, 0, 3);
    }

    // S = Q K^T (16x64 per wave); Q already carries the softmax scale
    f32x8 s[4];
#pragma unroll
    for (int ni = 0; ni < 4; ni++) s[ni] = zero8();
#pragma unroll
    for (int c = 0; c < 2; c++)
#pragma unroll
      for (int ni = 0; ni < 4; ni++) {
        bf16x16 kf = ldfrag(&Ks[ni * 16 + l15][c * 32 + kh * 8],
                            &Ks[ni * 16 + l15][c * 32 + 16 + kh * 8]);
        s[ni] = wmma_bf(qf[c], kf, s[ni]);
      }

    // online softmax: row stats across the 16 lanes holding each row (DPP)
    float corr[8];
#pragma unroll
    for (int r = 0; r < 8; r++) {
      float t = fmaxf(fmaxf(s[0][r], s[1][r]), fmaxf(s[2][r], s[3][r]));
      t = red16_max(t);
      float mn = fmaxf(mi[r], t);
      corr[r] = __expf(mi[r] - mn);
      li[r] *= corr[r];
      mi[r] = mn;
    }
#pragma unroll
    for (int ni = 0; ni < 4; ni++)
#pragma unroll
      for (int r = 0; r < 8; r++) oacc[ni][r] *= corr[r];

#pragma unroll
    for (int r = 0; r < 8; r++) {
      float rs = 0.f;
#pragma unroll
      for (int ni = 0; ni < 4; ni++) {
        float p = __expf(s[ni][r] - mi[r]);
        s[ni][r] = p;
        rs += p;
      }
      li[r] += red16_add(rs);
    }

    // P: C layout -> LDS (wave-private) -> A layout; then O += P V
#pragma unroll
    for (int ni = 0; ni < 4; ni++)
#pragma unroll
      for (int r = 0; r < 8; r++)
        Ps[w][kh * 8 + r][ni * 16 + l15] = f2bf(s[ni][r]);
    // same-wave DS ordering guarantees store->load visibility here
#pragma unroll
    for (int c = 0; c < 2; c++) {
      bf16x16 pf = ldfrag(&Ps[w][l15][c * 32 + kh * 8],
                          &Ps[w][l15][c * 32 + 16 + kh * 8]);
#pragma unroll
      for (int ni = 0; ni < 4; ni++) {
        bf16x16 vf = ldfrag(&Vt[ni * 16 + l15][c * 32 + kh * 8],
                            &Vt[ni * 16 + l15][c * 32 + 16 + kh * 8]);
        oacc[ni] = wmma_bf(pf, vf, oacc[ni]);
      }
    }
    __syncthreads();  // protect Ks/Vt restage
  }

  float inv[8];
#pragma unroll
  for (int r = 0; r < 8; r++) inv[r] = 1.f / li[r];
#pragma unroll
  for (int ni = 0; ni < 4; ni++)
#pragma unroll
    for (int r = 0; r < 8; r++) {
      int t = qt0 + kh * 8 + r;
      o[((size_t)(bb * T_ + t) * H_ + hh) * HS_ + ni * 16 + l15] =
          f2bf(oacc[ni][r] * inv[r]);
    }
}

// ------------------------------------------------- head-0 attention map out
// Q is pre-scaled, so raw dot products are already the softmax logits.
__global__ __launch_bounds__(128) void k_attn_map_h0(const u16* __restrict__ q,
                                                     const u16* __restrict__ k,
                                                     float* __restrict__ out) {
  __shared__ float qs[4][64];
  const int tid = threadIdx.x, lane = tid & 31, w = tid >> 5;
  const int row = blockIdx.x * 4 + w;  // b*T + t
  const int b = row >> 10;
  const u16* qp = q + (size_t)row * H_ * HS_;  // head 0
  for (int i = lane; i < 64; i += 32) qs[w][i] = bf2f(qp[i]);
  __syncthreads();

  float s[32];
  float mx = -3.0e38f;
  for (int jj = 0; jj < 32; jj++) {
    int j = jj * 32 + lane;
    const uint2* kp = (const uint2*)(k + (size_t)(b * T_ + j) * H_ * HS_);
    float acc = 0.f;
#pragma unroll
    for (int dd = 0; dd < 16; ++dd) {
      uint2 u = kp[dd];
      acc = fmaf(qs[w][dd * 4 + 0], bf2f((u16)(u.x & 0xFFFFu)), acc);
      acc = fmaf(qs[w][dd * 4 + 1], bf2f((u16)(u.x >> 16)), acc);
      acc = fmaf(qs[w][dd * 4 + 2], bf2f((u16)(u.y & 0xFFFFu)), acc);
      acc = fmaf(qs[w][dd * 4 + 3], bf2f((u16)(u.y >> 16)), acc);
    }
    s[jj] = acc;
    mx = fmaxf(mx, acc);
  }
#pragma unroll
  for (int off = 16; off; off >>= 1) mx = fmaxf(mx, __shfl_xor(mx, off, 32));
  float sum = 0.f;
  for (int jj = 0; jj < 32; jj++) { s[jj] = __expf(s[jj] - mx); sum += s[jj]; }
#pragma unroll
  for (int off = 16; off; off >>= 1) sum += __shfl_xor(sum, off, 32);
  float inv = 1.f / sum;
  float* orow = out + (size_t)row * T_;
  for (int jj = 0; jj < 32; jj++) orow[jj * 32 + lane] = s[jj] * inv;
}

// ------------------------------------------------------------- head kernels
__global__ __launch_bounds__(256) void k_meanpool(const float* __restrict__ h,
                                                  float* __restrict__ pooled) {
  int idx = blockIdx.x * 256 + threadIdx.x;  // b*D + d
  int b = idx >> 10, d = idx & 1023;
  float s = 0.f;
  for (int t = 0; t < T_; t++) s += h[((size_t)b * T_ + t) * D_ + d];
  pooled[idx] = s * (1.f / T_);
}

template <bool RELU>
__global__ __launch_bounds__(256) void k_fc(const float* __restrict__ in,
                                            const float* __restrict__ W,
                                            const float* __restrict__ bias,
                                            float* __restrict__ out,
                                            int M, int N, int K) {
  int idx = blockIdx.x * 256 + threadIdx.x;
  if (idx >= M * N) return;
  int m = idx / N, n = idx - m * N;
  float acc = bias[n];
  const float* ip = in + (size_t)m * K;
  for (int kk = 0; kk < K; ++kk) acc = fmaf(ip[kk], W[(size_t)kk * N + n], acc);
  if (RELU) acc = fmaxf(acc, 0.f);
  out[idx] = acc;
}

// ------------------------------------------------------------------- driver
extern "C" void kernel_launch(void* const* d_in, const int* in_sizes, int n_in,
                              void* d_out, int out_size, void* d_ws, size_t ws_size,
                              hipStream_t stream) {
  (void)in_sizes; (void)n_in; (void)out_size; (void)ws_size;
  const int*   x    = (const int*)d_in[0];
  const float* emb  = (const float*)d_in[1];
  const float* pos  = (const float*)d_in[2];
  const float* Wq   = (const float*)d_in[3];
  const float* Wk   = (const float*)d_in[4];
  const float* Wv   = (const float*)d_in[5];
  const float* Wp   = (const float*)d_in[6];
  const float* bp   = (const float*)d_in[7];
  const float* W1   = (const float*)d_in[8];
  const float* b1   = (const float*)d_in[9];
  const float* W2   = (const float*)d_in[10];
  const float* b2   = (const float*)d_in[11];
  const float* ln1g = (const float*)d_in[12];
  const float* ln1b = (const float*)d_in[13];
  const float* ln2g = (const float*)d_in[14];
  const float* ln2b = (const float*)d_in[15];
  const float* Wh   = (const float*)d_in[16];
  const float* bh   = (const float*)d_in[17];
  const float* Wo   = (const float*)d_in[18];
  const float* bo   = (const float*)d_in[19];

  char* base = (char*)d_ws;
  size_t off = 0;
  auto alloc = [&](size_t bytes) -> char* {
    char* p = base + off;
    off += (bytes + 255) & ~(size_t)255;
    return p;
  };
  const size_t DD = (size_t)D_ * D_, DF = (size_t)D_ * FF_;
  const int M = B_ * T_;

  u16* wq_b = (u16*)alloc((size_t)L_ * DD * 2);
  u16* wk_b = (u16*)alloc((size_t)L_ * DD * 2);
  u16* wv_b = (u16*)alloc((size_t)L_ * DD * 2);
  u16* wp_b = (u16*)alloc((size_t)L_ * DD * 2);
  u16* w1_b = (u16*)alloc((size_t)L_ * DF * 2);
  u16* w2_b = (u16*)alloc((size_t)L_ * DF * 2);
  float* h  = (float*)alloc((size_t)M * D_ * 4);
  u16* z    = (u16*)alloc((size_t)M * D_ * 2);
  u16* qb   = (u16*)alloc((size_t)M * D_ * 2);
  u16* kb   = (u16*)alloc((size_t)M * D_ * 2);
  u16* vb   = (u16*)alloc((size_t)M * D_ * 2);
  u16* ob   = (u16*)alloc((size_t)M * D_ * 2);
  u16* a1   = (u16*)alloc((size_t)M * FF_ * 2);
  float* pooled = (float*)alloc((size_t)B_ * D_ * 4);
  float* hid    = (float*)alloc((size_t)B_ * NH_ * 4);

  auto conv = [&](const float* s, u16* d, size_t n) {
    k_f32_to_bf16<<<dim3((unsigned)((n + 255) / 256)), 256, 0, stream>>>(s, d, n);
  };
  conv(Wq, wq_b, (size_t)L_ * DD);
  conv(Wk, wk_b, (size_t)L_ * DD);
  conv(Wv, wv_b, (size_t)L_ * DD);
  conv(Wp, wp_b, (size_t)L_ * DD);
  conv(W1, w1_b, (size_t)L_ * DF);
  conv(W2, w2_b, (size_t)L_ * DF);

  k_embed<<<M, 256, 0, stream>>>(x, emb, pos, h);

  const float scale = 0.03125f;  // D^-0.5, D=1024 (exact in bf16)
  const dim3 gD(D_ / 128, M / 128);    // N=1024 GEMMs
  const dim3 gF(FF_ / 128, M / 128);   // N=4096 GEMM

  for (int l = 0; l < L_; l++) {
    k_layernorm<<<M, 256, 0, stream>>>(h, ln1g + (size_t)l * D_,
                                       ln1b + (size_t)l * D_, z);
    // Q gets the softmax scale folded into its epilogue
    k_gemm<false, false, false, true, true><<<gD, 256, 0, stream>>>(
        z, wq_b + (size_t)l * DD, nullptr, nullptr, qb, M, D_, D_, scale);
    k_gemm<false, false, false, true, false><<<gD, 256, 0, stream>>>(
        z, wk_b + (size_t)l * DD, nullptr, nullptr, kb, M, D_, D_, 1.f);
    k_gemm<false, false, false, true, false><<<gD, 256, 0, stream>>>(
        z, wv_b + (size_t)l * DD, nullptr, nullptr, vb, M, D_, D_, 1.f);

    k_flash_attn<<<dim3(T_ / 128, H_, B_), 256, 0, stream>>>(qb, kb, vb, ob);

    float* amap = (float*)d_out + 8000 + (size_t)l * B_ * T_ * T_;
    k_attn_map_h0<<<M / 4, 128, 0, stream>>>(qb, kb, amap);

    k_gemm<true, false, true, false, false><<<gD, 256, 0, stream>>>(
        ob, wp_b + (size_t)l * DD, bp + (size_t)l * D_, h, h, M, D_, D_, 1.f);

    k_layernorm<<<M, 256, 0, stream>>>(h, ln2g + (size_t)l * D_,
                                       ln2b + (size_t)l * D_, z);
    k_gemm<true, true, false, true, false><<<gF, 256, 0, stream>>>(
        z, w1_b + (size_t)l * DF, b1 + (size_t)l * FF_, nullptr, a1, M, FF_, D_, 1.f);
    k_gemm<true, false, true, false, false><<<gD, 256, 0, stream>>>(
        a1, w2_b + (size_t)l * DF, b2 + (size_t)l * D_, h, h, M, D_, FF_, 1.f);
  }

  k_meanpool<<<(B_ * D_) / 256, 256, 0, stream>>>(h, pooled);
  k_fc<true><<<(B_ * NH_ + 255) / 256, 256, 0, stream>>>(pooled, Wh, bh, hid,
                                                         B_, NH_, D_);
  k_fc<false><<<(B_ * NO_ + 255) / 256, 256, 0, stream>>>(hid, Wo, bo,
                                                          (float*)d_out,
                                                          B_, NO_, NH_);
}
